// VectorQuantizer_10307921510619
// MI455X (gfx1250) — matrix operand order repaired and checked by reading, not compiled, not run on hardware
//
#include <hip/hip_runtime.h>

typedef __attribute__((ext_vector_type(16))) _Float16 v16h;
typedef __attribute__((ext_vector_type(8)))  _Float16 v8h;
typedef __attribute__((ext_vector_type(8)))  float    v8f;

#define M_ROWS          65536
#define D_DIM           64
#define K_CODES         8192
#define CHUNK           64
#define NCHUNKS         (K_CODES / CHUNK)          // 128
#define ROWS_PER_BLOCK  128
#define NBLOCKS         (M_ROWS / ROWS_PER_BLOCK)  // 512
#define ROW_PAD         72                         // 144B row stride in LDS
#define LOSS_SCALE      (1.25f / 4194304.0f)       // (1+beta)/(M*D)

// d_ws layout (bytes)
#define WS_HI_OFF    0u
#define WS_LO_OFF    (1u << 20)             // 524288 halfs = 1MB each plane
#define WS_ESQ_OFF   (2u << 20)             // 8192 f32
#define WS_PART_OFF  ((2u << 20) + 32768u)  // 512 f32

static __device__ __forceinline__ int lane_id() {
  return (int)__builtin_amdgcn_mbcnt_lo(~0u, 0u);   // wave32: 0..31
}
static __device__ __forceinline__ int bperm_i(int src_lane, int v) {
  return __builtin_amdgcn_ds_bpermute(src_lane << 2, v);
}
static __device__ __forceinline__ float bperm_f(int src_lane, float v) {
  return __builtin_bit_cast(float,
      __builtin_amdgcn_ds_bpermute(src_lane << 2, __builtin_bit_cast(int, v)));
}
// 16 contiguous halfs (16B-aligned) -> v16h, as two ds_load_b128
static __device__ __forceinline__ v16h load16h(const _Float16* p) {
  v8h a = *(const v8h*)p;
  v8h b = *(const v8h*)(p + 8);
  return __builtin_shufflevector(a, b, 0, 1, 2, 3, 4, 5, 6, 7,
                                 8, 9, 10, 11, 12, 13, 14, 15);
}

// ---------------------------------------------------------------------------
// Prep 1: split codebook f32 -> hi/lo f16 planes in d_ws (once per launch)
// ---------------------------------------------------------------------------
__global__ __launch_bounds__(256) void vq_prep_split(const float* __restrict__ cb,
                                                     _Float16* __restrict__ ws_hi,
                                                     _Float16* __restrict__ ws_lo) {
  int gid = blockIdx.x * 256 + threadIdx.x;      // 131072 threads, 4 elems each
  float4 v = ((const float4*)cb)[gid];
  int i0 = gid * 4;
  float f[4] = {v.x, v.y, v.z, v.w};
#pragma unroll
  for (int j = 0; j < 4; ++j) {
    _Float16 hi = (_Float16)f[j];
    ws_hi[i0 + j] = hi;
    ws_lo[i0 + j] = (_Float16)(f[j] - (float)hi);
  }
}

// ---------------------------------------------------------------------------
// Prep 2: per-code squared norms in f32
// ---------------------------------------------------------------------------
__global__ __launch_bounds__(256) void vq_prep_esq(const float* __restrict__ cb,
                                                   float* __restrict__ ws_esq) {
  int code = blockIdx.x * 256 + threadIdx.x;     // 8192 threads
  const float4* r = (const float4*)(cb + (size_t)code * D_DIM);
  float s = 0.f;
#pragma unroll
  for (int i = 0; i < 16; ++i) {
    float4 q = r[i];
    s += q.x * q.x + q.y * q.y + q.z * q.z + q.w * q.w;
  }
  ws_esq[code] = s;
}

// ---------------------------------------------------------------------------
// Main: pairwise-distance GEMM via split-f16 WMMA + argmin + gather
// ---------------------------------------------------------------------------
__global__ __launch_bounds__(256) void vq_main(const float* __restrict__ z,
                                               const float* __restrict__ cb,
                                               const _Float16* __restrict__ ws_hi,
                                               const _Float16* __restrict__ ws_lo,
                                               const float* __restrict__ ws_esq,
                                               float* __restrict__ out_q,
                                               float* __restrict__ out_misc,
                                               float* __restrict__ partial) {
  __shared__ _Float16 lds_hi[CHUNK * ROW_PAD];   // 9216B, padded rows
  __shared__ _Float16 lds_lo[CHUNK * ROW_PAD];
  __shared__ float    lds_esq[CHUNK];
  __shared__ float    lds_wsum[8];

  const int tid    = threadIdx.x;
  const int lane   = lane_id();
  const int wv     = tid >> 5;
  const int hilane = (lane >= 16) ? 1 : 0;
  const int mrow   = lane & 15;
  const int rowbase = blockIdx.x * ROWS_PER_BLOCK + wv * 16;

  // ---- A fragments: 16 z rows per wave, f16 hi/lo split.
  // A k order used here (contiguous, same as B): m = lane%16, k(j) = j + 16*hilane (+32h)
  v16h a_hi[2], a_lo[2];
  {
    const float4* zr4 = (const float4*)(z + (size_t)(rowbase + mrow) * D_DIM);
#pragma unroll
    for (int h = 0; h < 2; ++h)
#pragma unroll
      for (int rr = 0; rr < 2; ++rr) {
        int q0 = rr * 2 + hilane * 4 + h * 8;    // float4 index of 8-float run
        float4 f0 = zr4[q0];
        float4 f1 = zr4[q0 + 1];
        float vals[8] = {f0.x, f0.y, f0.z, f0.w, f1.x, f1.y, f1.z, f1.w};
#pragma unroll
        for (int j2 = 0; j2 < 8; ++j2) {
          float v = vals[j2];
          _Float16 hi = (_Float16)v;
          a_hi[h][rr * 8 + j2] = hi;
          a_lo[h][rr * 8 + j2] = (_Float16)(v - (float)hi);
        }
      }
  }

  float bestv[8];
  int   besti[8];
#pragma unroll
  for (int r = 0; r < 8; ++r) { bestv[r] = 3.4e38f; besti[r] = 0; }

  for (int c = 0; c < NCHUNKS; ++c) {
    const uint4* src_hi = (const uint4*)(ws_hi + (size_t)c * (CHUNK * D_DIM));
    const uint4* src_lo = (const uint4*)(ws_lo + (size_t)c * (CHUNK * D_DIM));
    // Stage chunk: pure b128 copies into padded-row LDS planes
#pragma unroll
    for (int i = 0; i < 2; ++i) {
      int q = tid + i * 256;                  // 512 x 16B per plane
      int code = q >> 3, dd = (q & 7) << 3;   // dim base (multiple of 8)
      *(uint4*)&lds_hi[code * ROW_PAD + dd] = src_hi[q];
      *(uint4*)&lds_lo[code * ROW_PAD + dd] = src_lo[q];
    }
    if (tid < CHUNK) lds_esq[tid] = ws_esq[c * CHUNK + tid];
    if (c + 1 < NCHUNKS) {                    // global_prefetch_b8 next chunk
      __builtin_prefetch((const char*)(src_hi + 512) + tid * 32, 0, 1);
      __builtin_prefetch((const char*)(src_lo + 512) + tid * 32, 0, 1);
    }
    __syncthreads();

    // 4 tiles of 16 codes; 6 WMMAs each: hi*hi + lo*hi + hi*lo, f32 accum
    for (int t = 0; t < 4; ++t) {
      int n = (t << 4) | mrow;                // code column for this lane
      // B k order used here (contiguous, same as A): n = lane%16, k(j) = j + 16*hilane (+32h)
      int base = n * ROW_PAD + hilane * 16;
      v16h bh0 = load16h(&lds_hi[base]);
      v16h bh1 = load16h(&lds_hi[base + 32]);
      v16h bl0 = load16h(&lds_lo[base]);
      v16h bl1 = load16h(&lds_lo[base + 32]);

      v8f acc = {};
      acc = __builtin_amdgcn_wmma_f32_16x16x32_f16(false, a_hi[0], false, bh0, (short)0, acc, false, false);
      acc = __builtin_amdgcn_wmma_f32_16x16x32_f16(false, a_hi[1], false, bh1, (short)0, acc, false, false);
      acc = __builtin_amdgcn_wmma_f32_16x16x32_f16(false, a_lo[0], false, bh0, (short)0, acc, false, false);
      acc = __builtin_amdgcn_wmma_f32_16x16x32_f16(false, a_lo[1], false, bh1, (short)0, acc, false, false);
      acc = __builtin_amdgcn_wmma_f32_16x16x32_f16(false, a_hi[0], false, bl0, (short)0, acc, false, false);
      acc = __builtin_amdgcn_wmma_f32_16x16x32_f16(false, a_hi[1], false, bl1, (short)0, acc, false, false);

      // argmin surrogate: ||e||^2 - 2*z.e   (z^2 constant per row)
      float esq  = lds_esq[n];
      int   gidx = (c << 6) | n;              // strictly increasing per lane
#pragma unroll
      for (int r = 0; r < 8; ++r) {
        float dv = __builtin_fmaf(-2.0f, acc[r], esq);
        if (dv < bestv[r]) { bestv[r] = dv; besti[r] = gidx; }  // first-hit kept
      }
    }
    __syncthreads();
  }

  // Cross-lane argmin within each 16-lane half (lanes 0-15: rows r..r+7,
  // lanes 16-31: rows r+8..r+15); tie-break on lower index = first occurrence.
#pragma unroll
  for (int off = 1; off < 16; off <<= 1) {
#pragma unroll
    for (int r = 0; r < 8; ++r) {
      float ov = bperm_f(lane ^ off, bestv[r]);
      int   oi = bperm_i(lane ^ off, besti[r]);
      if (ov < bestv[r] || (ov == bestv[r] && oi < besti[r])) {
        bestv[r] = ov; besti[r] = oi;
      }
    }
  }

  if ((lane & 15) == 0) {
    int rbase = rowbase + (hilane ? 8 : 0);
#pragma unroll
    for (int r = 0; r < 8; ++r)
      out_misc[1 + rbase + r] = (float)besti[r];
  }

  // Gather exact f32 codebook rows + accumulate sum((q - z)^2)
  float lsum = 0.f;
  for (int r = 0; r < 16; ++r) {
    int idx = (r < 8) ? bperm_i(0, besti[r]) : bperm_i(16, besti[r - 8]);
    int row = rowbase + r;
    float2 q  = ((const float2*)(cb + (size_t)idx * D_DIM))[lane];
    float2 zz = ((const float2*)(z  + (size_t)row * D_DIM))[lane];
    ((float2*)(out_q + (size_t)row * D_DIM))[lane] = q;
    float dx = q.x - zz.x, dy = q.y - zz.y;
    lsum += dx * dx + dy * dy;
  }
#pragma unroll
  for (int off = 1; off < 32; off <<= 1) lsum += bperm_f(lane ^ off, lsum);
  if (lane == 0) lds_wsum[wv] = lsum;
  __syncthreads();
  if (tid == 0) {
    float s = 0.f;
#pragma unroll
    for (int w = 0; w < 8; ++w) s += lds_wsum[w];
    partial[blockIdx.x] = s;                   // deterministic per-block partial
  }
}

// ---------------------------------------------------------------------------
// Deterministic fixed-order reduction of 512 block partials -> vq_loss
// ---------------------------------------------------------------------------
__global__ __launch_bounds__(256) void vq_reduce(const float* __restrict__ partial,
                                                 float* __restrict__ out_misc) {
  __shared__ float sm[256];
  float s = partial[threadIdx.x] + partial[threadIdx.x + 256];
  sm[threadIdx.x] = s;
  __syncthreads();
  for (int off = 128; off > 0; off >>= 1) {
    if ((int)threadIdx.x < off) sm[threadIdx.x] += sm[threadIdx.x + off];
    __syncthreads();
  }
  if (threadIdx.x == 0) out_misc[0] = sm[0] * LOSS_SCALE;
}

extern "C" void kernel_launch(void* const* d_in, const int* in_sizes, int n_in,
                              void* d_out, int out_size, void* d_ws, size_t ws_size,
                              hipStream_t stream) {
  (void)in_sizes; (void)n_in; (void)out_size; (void)ws_size;
  const float* z  = (const float*)d_in[0];        // (16,4096,64) f32
  const float* cb = (const float*)d_in[1];        // (8192,64)    f32
  float* out      = (float*)d_out;
  float* out_q    = out;                          // quantised (M*D f32)
  float* out_misc = out + (size_t)M_ROWS * D_DIM; // [0]=loss, [1..]=indices

  char* ws = (char*)d_ws;                         // needs ~2.1MB scratch
  _Float16* ws_hi  = (_Float16*)(ws + WS_HI_OFF);
  _Float16* ws_lo  = (_Float16*)(ws + WS_LO_OFF);
  float*    ws_esq = (float*)(ws + WS_ESQ_OFF);
  float*    partial = (float*)(ws + WS_PART_OFF);

  vq_prep_split<<<K_CODES * D_DIM / 4 / 256, 256, 0, stream>>>(cb, ws_hi, ws_lo);
  vq_prep_esq  <<<K_CODES / 256,             256, 0, stream>>>(cb, ws_esq);
  vq_main      <<<NBLOCKS, 256, 0, stream>>>(z, cb, ws_hi, ws_lo, ws_esq,
                                             out_q, out_misc, partial);
  vq_reduce    <<<1, 256, 0, stream>>>(partial, out_misc);
}
